// PointConv_45981919871336
// MI455X (gfx1250) — matrix operand order repaired and checked by reading, not compiled, not run on hardware
//
#include <hip/hip_runtime.h>
#include <hip/hip_bf16.h>

// ============================================================================
// PointConv pipeline for MI455X (gfx1250, wave32, WMMA bf16).
//
// Dominant cost: local_nn over 340k message rows (~0.39 TFLOP). Strategy:
// fully-fused per-32-row megakernels keeping all [rows,512] intermediates in
// LDS; bf16 WMMA (v_wmma_f32_16x16x32_bf16) with f32 accumulation; weights
// converted once to bf16 (L2-resident, ~1.9MB); segment_sum via
// global_atomic_add_f32 into a 41MB agg buffer (fits L2).
// 32 rows/block halves L2 weight traffic vs 16 rows (B-fragments reused
// across two row-tiles per load): ~12.5GB L2 reads for local_nn.
//
// d_in layout assumption: top-level dict insertion order (x, pos, edge_index,
// batch, batch_size, params), params flattened JAX-pytree style (keys sorted
// recursively): dec, enc, enc_pos, gate_nn, global_nn, local_nn; within a
// bias block: W0,W1,W2,b0,b1,b2,g0b,g0w,g1b,g1w; without bias:
// W0,W1,W2,b2,g0b,g0w,g1b,g1w.
// ============================================================================

typedef unsigned short u16;
typedef __attribute__((ext_vector_type(16))) __bf16 v16bf;
typedef __attribute__((ext_vector_type(8)))  float  v8f;

union FragU { uint4 q[2]; v16bf v; };

__device__ __forceinline__ u16 f32_to_bf16(float f) {
  unsigned int u = __float_as_uint(f);
  unsigned int r = u + 0x7FFFu + ((u >> 16) & 1u);   // round-to-nearest-even
  return (u16)(r >> 16);
}
__device__ __forceinline__ float bf16_to_f32(u16 h) {
  return __uint_as_float(((unsigned int)h) << 16);
}
__device__ __forceinline__ void atomicMaxFloat(float* addr, float val) {
  if (val >= 0.f) atomicMax((int*)addr, __float_as_int(val));
  else            atomicMin((unsigned int*)addr, __float_as_uint(val));
}

// ---------------------------------------------------------------------------
// WMMA fragment loaders per CDNA5 ISA layouts (16-bit A 16x32, B 32x16).
// A: lane<16 -> M=lane, K = kb+0..7 and kb+16..23; lane>=16 -> K = +8 shift.
// B: lane gives N=lane&15; same K order as A: lane<16 K=kb+0..7 and kb+16..23,
//    lane>=16 -> K = +8 shift (weights stored [N][K] row-major).
// ---------------------------------------------------------------------------
__device__ __forceinline__ v16bf load_a_frag(const u16* A, int lda, int kb, int lane) {
  int half = lane >> 4, m = lane & 15;
  const u16* p = A + m * lda + kb + 8 * half;
  FragU f;
  f.q[0] = *reinterpret_cast<const uint4*>(p);
  f.q[1] = *reinterpret_cast<const uint4*>(p + 16);
  return f.v;
}
__device__ __forceinline__ v16bf load_b_frag(const u16* Wt, int ldk, int n0, int kb, int lane) {
  int half = lane >> 4, nn = lane & 15;
  const u16* p = Wt + (long)(n0 + nn) * ldk + kb + 8 * half;
  FragU f;
  f.q[0] = *reinterpret_cast<const uint4*>(p);
  f.q[1] = *reinterpret_cast<const uint4*>(p + 16);
  return f.v;
}

// R-row GEMM: sOut[R][NCOLS] (f32, +optional bias) = sA[R][K] x Wt^T.
// 8 waves x (NCOLS/128) col-tiles each, R/16 row-tiles; each B fragment is
// loaded once per (kb, col-tile) and reused across all row-tiles.
// C layout: vgpr r -> row rt*16 + r + 8*half, col (wave*TILES+t)*16 + ln.
template <int R, int NCOLS, int K>
__device__ __forceinline__ void gemm_r(const u16* sA, int lda, const u16* Wt,
                                       float* sOut, const float* bias, int tid) {
  constexpr int TILES = NCOLS / 128;   // col tiles per wave (8 waves * 16 cols)
  constexpr int RT    = R / 16;        // row tiles
  int wave = tid >> 5, lane = tid & 31, half = lane >> 4, ln = lane & 15;
  v8f acc[RT][TILES];
  for (int rt = 0; rt < RT; ++rt)
    for (int t = 0; t < TILES; ++t)
      for (int r = 0; r < 8; ++r) acc[rt][t][r] = 0.f;
  for (int kb = 0; kb < K; kb += 32) {
    v16bf a[RT];
    for (int rt = 0; rt < RT; ++rt)
      a[rt] = load_a_frag(sA + rt * 16 * lda, lda, kb, lane);
    for (int t = 0; t < TILES; ++t) {
      int n0 = (wave * TILES + t) * 16;
      v16bf b = load_b_frag(Wt, K, n0, kb, lane);
      for (int rt = 0; rt < RT; ++rt)
        acc[rt][t] = __builtin_amdgcn_wmma_f32_16x16x32_bf16(
            false, a[rt], false, b, (short)0, acc[rt][t], false, false);
    }
  }
  for (int rt = 0; rt < RT; ++rt)
    for (int t = 0; t < TILES; ++t) {
      int col = (wave * TILES + t) * 16 + ln;
      float bz = bias ? bias[col] : 0.f;
      for (int r = 0; r < 8; ++r)
        sOut[(rt * 16 + r + 8 * half) * NCOLS + col] = acc[rt][t][r] + bz;
    }
}

// GroupNorm (per-row, G groups) + ReLU + convert to bf16, R rows x C cols.
template <int R, int C, int G>
__device__ __forceinline__ void gn_relu_to_bf16(const float* sIn, u16* sOut,
                                                const float* gw, const float* gb,
                                                float* sMean, float* sRstd, int tid) {
  constexpr int GS = C / G;
  __syncthreads();
  for (int p = tid; p < R * G; p += 256) {
    int i = p / G, g = p % G;
    const float* row = sIn + i * C + g * GS;
    float s = 0.f, s2 = 0.f;
    for (int k = 0; k < GS; ++k) { float v = row[k]; s += v; s2 += v * v; }
    float mu = s * (1.f / GS);
    float var = s2 * (1.f / GS) - mu * mu;
    sMean[p] = mu;
    sRstd[p] = rsqrtf(var + 1e-5f);
  }
  __syncthreads();
  for (int idx = tid; idx < R * C; idx += 256) {
    int i = idx / C, c = idx % C, g = c / GS;
    float v = (sIn[idx] - sMean[i * G + g]) * sRstd[i * G + g] * gw[c] + gb[c];
    sOut[idx] = f32_to_bf16(v > 0.f ? v : 0.f);
  }
  __syncthreads();
}

// ---------------------------------------------------------------------------
// K0: f32 -> bf16 weight conversion with K-padding (local W0: 100 -> 128).
// ---------------------------------------------------------------------------
__global__ void convert_w_kernel(const float* __restrict__ src, u16* __restrict__ dst,
                                 int rows, int cs, int cd) {
  long idx = (long)blockIdx.x * 256 + threadIdx.x;
  if (idx >= (long)rows * cd) return;
  int r = (int)(idx / cd), c = (int)(idx % cd);
  float v = (c < cs) ? src[(long)r * cs + c] : 0.f;
  dst[idx] = f32_to_bf16(v);
}

// ---------------------------------------------------------------------------
// K1: node encoders (4->32 and 4->64 MLP blocks), one thread per node (VALU;
// K=4 is far below WMMA's K=32 granularity). Writes feat as bf16 [N][96].
// ---------------------------------------------------------------------------
template <int DH, int G>
__device__ __forceinline__ void gn_relu_serial(float* h, const float* gw, const float* gb) {
  constexpr int GS = DH / G;
  for (int g = 0; g < G; ++g) {
    float s = 0.f, s2 = 0.f;
    for (int k = 0; k < GS; ++k) { float v = h[g * GS + k]; s += v; s2 += v * v; }
    float mu = s * (1.f / GS);
    float var = s2 * (1.f / GS) - mu * mu;
    float rs = rsqrtf(var + 1e-5f);
    for (int k = 0; k < GS; ++k) {
      int c = g * GS + k;
      float v = (h[c] - mu) * rs * gw[c] + gb[c];
      h[c] = v > 0.f ? v : 0.f;
    }
  }
}

template <int DIN, int DH, int G>
__device__ __forceinline__ void block_small(const float* in, float* out,
    const float* W0, const float* b0, const float* g0w, const float* g0b,
    const float* W1, const float* b1, const float* g1w, const float* g1b,
    const float* W2, const float* b2) {
  float h[DH], t[DH];
  for (int o = 0; o < DH; ++o) {
    float a = b0[o];
    for (int k = 0; k < DIN; ++k) a += W0[o * DIN + k] * in[k];
    h[o] = a;
  }
  gn_relu_serial<DH, G>(h, g0w, g0b);
  for (int o = 0; o < DH; ++o) {
    float a = b1[o];
    for (int k = 0; k < DH; ++k) a += W1[o * DH + k] * h[k];
    t[o] = a;
  }
  gn_relu_serial<DH, G>(t, g1w, g1b);
  for (int o = 0; o < DH; ++o) {
    float a = b2[o];
    for (int k = 0; k < DH; ++k) a += W2[o * DH + k] * t[k];
    out[o] = a;
  }
}

struct EncParams {
  const float *x, *pos;
  const float *eW0, *eW1, *eW2, *eb0, *eb1, *eb2, *eg0w, *eg0b, *eg1w, *eg1b;
  const float *pW0, *pW1, *pW2, *pb0, *pb1, *pb2, *pg0w, *pg0b, *pg1w, *pg1b;
  u16* featbf;
  int N;
};

__global__ __launch_bounds__(256) void encode_kernel(EncParams P) {
  int n = blockIdx.x * 256 + threadIdx.x;
  if (n >= P.N) return;
  float in[4], pin[4];
  for (int k = 0; k < 4; ++k) { in[k] = P.x[(long)n * 4 + k]; pin[k] = P.pos[(long)n * 4 + k]; }
  float h[32], hp[64];
  block_small<4, 32, 4>(in, h, P.eW0, P.eb0, P.eg0w, P.eg0b,
                        P.eW1, P.eb1, P.eg1w, P.eg1b, P.eW2, P.eb2);
  block_small<4, 64, 4>(pin, hp, P.pW0, P.pb0, P.pg0w, P.pg0b,
                        P.pW1, P.pb1, P.pg1w, P.pg1b, P.pW2, P.pb2);
  u16* dst = P.featbf + (long)n * 96;
  for (int c = 0; c < 32; ++c) dst[c] = f32_to_bf16(h[c]);
  for (int c = 0; c < 64; ++c) dst[32 + c] = f32_to_bf16(hp[c]);
}

// ---------------------------------------------------------------------------
// K2: local_nn megakernel. 32 message rows per WG, fully fused in LDS:
// gather msg -> GEMM(128->512) -> GN -> GEMM(512->512) -> GN -> GEMM + b2
// -> atomic scatter-add into agg[dst].
// ---------------------------------------------------------------------------
struct LocalParams {
  const float* pos;
  const int* ei;
  const u16* featbf;
  const u16 *W0, *W1, *W2;
  const float *b2, *g0w, *g0b, *g1w, *g1b;
  float* agg;
  int N, E;
};

__global__ __launch_bounds__(256) void local_nn_kernel(LocalParams P) {
  constexpr int R = 32;
  __shared__ __align__(16) u16   sA[R * 128];        // 8 KB
  __shared__ __align__(16) float sStage[R * 512];    // 64 KB
  __shared__ __align__(16) u16   sH[R * 512];        // 32 KB
  __shared__ float sMean[R * 8], sRstd[R * 8];
  __shared__ int sSrc[R], sDst[R], sValid[R];
  int tid = threadIdx.x;
  long base = (long)blockIdx.x * R;
  long rows = (long)P.E + P.N;

  if (tid < R) {
    long r = base + tid;
    int s = 0, d = 0, valid = 0;
    if (r < rows) {
      valid = 1;
      if (r < P.E) { s = P.ei[r]; d = P.ei[(long)P.E + r]; }
      else { s = d = (int)(r - P.E); }
    }
    sSrc[tid] = s; sDst[tid] = d; sValid[tid] = valid;
  }
  __syncthreads();

  // Build message rows: [feat[src](96) | pos[src]-pos[dst](4) | 0-pad(28)]
  for (int idx = tid; idx < R * 128; idx += 256) {
    int i = idx >> 7, c = idx & 127;
    u16 v = 0;
    if (sValid[i]) {
      int s = sSrc[i], d = sDst[i];
      if (c < 96) v = P.featbf[(long)s * 96 + c];
      else if (c < 100) {
        int k = c - 96;
        v = f32_to_bf16(P.pos[(long)s * 4 + k] - P.pos[(long)d * 4 + k]);
      }
    }
    sA[idx] = v;
  }
  __syncthreads();

  gemm_r<R, 512, 128>(sA, 128, P.W0, sStage, nullptr, tid);
  gn_relu_to_bf16<R, 512, 8>(sStage, sH, P.g0w, P.g0b, sMean, sRstd, tid);
  gemm_r<R, 512, 512>(sH, 512, P.W1, sStage, nullptr, tid);
  gn_relu_to_bf16<R, 512, 8>(sStage, sH, P.g1w, P.g1b, sMean, sRstd, tid);
  gemm_r<R, 512, 512>(sH, 512, P.W2, sStage, P.b2, tid);
  __syncthreads();

  for (int idx = tid; idx < R * 512; idx += 256) {
    int i = idx >> 9, c = idx & 511;
    if (sValid[i]) atomicAdd(&P.agg[(long)sDst[i] * 512 + c], sStage[idx]);
  }
}

// ---------------------------------------------------------------------------
// K3: global_nn megakernel: agg[N,512] -> xg[N,256] (3 GEMMs + 2 GN in LDS).
// ---------------------------------------------------------------------------
struct GlobParams {
  const float* agg;
  const u16 *W0, *W1, *W2;
  const float *b2, *g0w, *g0b, *g1w, *g1b;
  float* xg;
  int N;
};

__global__ __launch_bounds__(256) void global_nn_kernel(GlobParams P) {
  constexpr int R = 32;
  __shared__ __align__(16) u16   sA[R * 512];        // 32 KB
  __shared__ __align__(16) float sStage[R * 256];    // 32 KB
  __shared__ __align__(16) u16   sH[R * 256];        // 16 KB
  __shared__ float sMean[R * 8], sRstd[R * 8];
  int tid = threadIdx.x;
  long base = (long)blockIdx.x * R;

  for (int idx = tid; idx < R * 512; idx += 256) {
    int i = idx >> 9, c = idx & 511;
    long row = base + i;
    float v = (row < P.N) ? P.agg[row * 512 + c] : 0.f;
    sA[idx] = f32_to_bf16(v);
  }
  __syncthreads();

  gemm_r<R, 256, 512>(sA, 512, P.W0, sStage, nullptr, tid);
  gn_relu_to_bf16<R, 256, 8>(sStage, sH, P.g0w, P.g0b, sMean, sRstd, tid);
  gemm_r<R, 256, 256>(sH, 256, P.W1, sStage, nullptr, tid);
  gn_relu_to_bf16<R, 256, 8>(sStage, sH, P.g1w, P.g1b, sMean, sRstd, tid);
  gemm_r<R, 256, 256>(sH, 256, P.W2, sStage, P.b2, tid);
  __syncthreads();

  for (int idx = tid; idx < R * 256; idx += 256) {
    int i = idx >> 8, c = idx & 255;
    long row = base + i;
    if (row < P.N) P.xg[row * 256 + c] = sStage[idx];
  }
}

// ---------------------------------------------------------------------------
// K4: gate_nn megakernel: xg[N,256] -> gate[N] (2 GEMMs + 2 GN + f32 dot).
// ---------------------------------------------------------------------------
struct GateParams {
  const float* xg;
  const u16 *W0, *W1;
  const float *W2f, *b2, *g0w, *g0b, *g1w, *g1b;
  float* gate;
  int N;
};

__global__ __launch_bounds__(256) void gate_nn_kernel(GateParams P) {
  constexpr int R = 32;
  __shared__ __align__(16) u16   sA[R * 256];        // 16 KB
  __shared__ __align__(16) float sStage[R * 256];    // 32 KB
  __shared__ __align__(16) u16   sH[R * 256];        // 16 KB
  __shared__ float sMean[R * 8], sRstd[R * 8];
  int tid = threadIdx.x;
  long base = (long)blockIdx.x * R;

  for (int idx = tid; idx < R * 256; idx += 256) {
    int i = idx >> 8, c = idx & 255;
    long row = base + i;
    float v = (row < P.N) ? P.xg[row * 256 + c] : 0.f;
    sA[idx] = f32_to_bf16(v);
  }
  __syncthreads();

  gemm_r<R, 256, 256>(sA, 256, P.W0, sStage, nullptr, tid);
  gn_relu_to_bf16<R, 256, 8>(sStage, sH, P.g0w, P.g0b, sMean, sRstd, tid);
  gemm_r<R, 256, 256>(sH, 256, P.W1, sStage, nullptr, tid);
  gn_relu_to_bf16<R, 256, 8>(sStage, sH, P.g1w, P.g1b, sMean, sRstd, tid);

  if (tid < R) {
    long row = base + tid;
    if (row < P.N) {
      float s = 0.f;
      for (int k = 0; k < 256; ++k) s += bf16_to_f32(sH[tid * 256 + k]) * P.W2f[k];
      P.gate[row] = s + P.b2[0];
    }
  }
}

// ---------------------------------------------------------------------------
// K5: GlobalAttention segment-softmax pooling.
// ---------------------------------------------------------------------------
__global__ void init_gmax_kernel(float* gmax) {
  if (threadIdx.x < 16) gmax[threadIdx.x] = -INFINITY;
}
__global__ void gate_max_kernel(const float* gate, const int* batch, float* gmax, int N) {
  int n = blockIdx.x * 256 + threadIdx.x;
  if (n >= N) return;
  atomicMaxFloat(&gmax[batch[n]], gate[n]);
}
__global__ void gate_exp_kernel(const float* gate, const int* batch, const float* gmax,
                                float* ebuf, float* denom, int N) {
  int n = blockIdx.x * 256 + threadIdx.x;
  if (n >= N) return;
  float e = expf(gate[n] - gmax[batch[n]]);
  ebuf[n] = e;
  atomicAdd(&denom[batch[n]], e);
}
__global__ void pool_kernel(const float* xg, const int* batch, const float* ebuf,
                            const float* denom, float* pooled, int N) {
  long idx = (long)blockIdx.x * 256 + threadIdx.x;
  if (idx >= (long)N * 256) return;
  int n = (int)(idx >> 8), c = (int)(idx & 255);
  int b = batch[n];
  float alpha = ebuf[n] / fmaxf(denom[b], 1e-16f);
  atomicAdd(&pooled[b * 256 + c], alpha * xg[idx]);
}

// ---------------------------------------------------------------------------
// K6: decoder block on pooled [16,256] -> out [16,6]. One WG, f32 VALU.
// ---------------------------------------------------------------------------
struct DecParams {
  const float* pooled;
  const float *W0, *W1, *W2, *b0, *b1, *b2, *g0w, *g0b, *g1w, *g1b;
  float* out;
};

__global__ __launch_bounds__(256) void dec_kernel(DecParams P) {
  __shared__ float sIn[256], sH[256], sM[8], sR[8];
  int tid = threadIdx.x;
  for (int row = 0; row < 16; ++row) {
    sIn[tid] = P.pooled[row * 256 + tid];
    __syncthreads();
    float a = P.b0[tid];
    for (int k = 0; k < 256; ++k) a += P.W0[tid * 256 + k] * sIn[k];
    sH[tid] = a;
    __syncthreads();
    if (tid < 8) {
      float s = 0.f, s2 = 0.f;
      for (int k = 0; k < 32; ++k) { float v = sH[tid * 32 + k]; s += v; s2 += v * v; }
      float mu = s * (1.f / 32.f), var = s2 * (1.f / 32.f) - mu * mu;
      sM[tid] = mu; sR[tid] = rsqrtf(var + 1e-5f);
    }
    __syncthreads();
    {
      int g = tid >> 5;
      float v = (sH[tid] - sM[g]) * sR[g] * P.g0w[tid] + P.g0b[tid];
      sIn[tid] = v > 0.f ? v : 0.f;
    }
    __syncthreads();
    float b = P.b1[tid];
    for (int k = 0; k < 256; ++k) b += P.W1[tid * 256 + k] * sIn[k];
    sH[tid] = b;
    __syncthreads();
    if (tid < 8) {
      float s = 0.f, s2 = 0.f;
      for (int k = 0; k < 32; ++k) { float v = sH[tid * 32 + k]; s += v; s2 += v * v; }
      float mu = s * (1.f / 32.f), var = s2 * (1.f / 32.f) - mu * mu;
      sM[tid] = mu; sR[tid] = rsqrtf(var + 1e-5f);
    }
    __syncthreads();
    {
      int g = tid >> 5;
      float v = (sH[tid] - sM[g]) * sR[g] * P.g1w[tid] + P.g1b[tid];
      sIn[tid] = v > 0.f ? v : 0.f;
    }
    __syncthreads();
    if (tid < 6) {
      float s = P.b2[tid];
      for (int k = 0; k < 256; ++k) s += P.W2[tid * 256 + k] * sIn[k];
      P.out[row * 6 + tid] = s;
    }
    __syncthreads();
  }
}

// ===========================================================================
// kernel_launch
// ===========================================================================
extern "C" void kernel_launch(void* const* d_in, const int* in_sizes, int n_in,
                              void* d_out, int out_size, void* d_ws, size_t ws_size,
                              hipStream_t stream) {
  const int N = in_sizes[0] / 4;   // 20000
  const int E = in_sizes[2] / 2;   // 320000
  const int B = 16;

  const float* x     = (const float*)d_in[0];
  const float* pos   = (const float*)d_in[1];
  const int*   ei    = (const int*)d_in[2];
  const int*   batch = (const int*)d_in[3];
  // d_in[4] = batch_size scalar (device); B hardcoded from reference setup.

  // params (pytree-sorted): dec[5..14], enc[15..24], enc_pos[25..34],
  // gate[35..42], global[43..50], local[51..58]
  const float* dW0 = (const float*)d_in[5];  const float* dW1 = (const float*)d_in[6];
  const float* dW2 = (const float*)d_in[7];  const float* db0 = (const float*)d_in[8];
  const float* db1 = (const float*)d_in[9];  const float* db2 = (const float*)d_in[10];
  const float* dg0b = (const float*)d_in[11]; const float* dg0w = (const float*)d_in[12];
  const float* dg1b = (const float*)d_in[13]; const float* dg1w = (const float*)d_in[14];

  const float* eW0 = (const float*)d_in[15]; const float* eW1 = (const float*)d_in[16];
  const float* eW2 = (const float*)d_in[17]; const float* eb0 = (const float*)d_in[18];
  const float* eb1 = (const float*)d_in[19]; const float* eb2 = (const float*)d_in[20];
  const float* eg0b = (const float*)d_in[21]; const float* eg0w = (const float*)d_in[22];
  const float* eg1b = (const float*)d_in[23]; const float* eg1w = (const float*)d_in[24];

  const float* pW0 = (const float*)d_in[25]; const float* pW1 = (const float*)d_in[26];
  const float* pW2 = (const float*)d_in[27]; const float* pb0 = (const float*)d_in[28];
  const float* pb1 = (const float*)d_in[29]; const float* pb2 = (const float*)d_in[30];
  const float* pg0b = (const float*)d_in[31]; const float* pg0w = (const float*)d_in[32];
  const float* pg1b = (const float*)d_in[33]; const float* pg1w = (const float*)d_in[34];

  const float* aW0 = (const float*)d_in[35]; const float* aW1 = (const float*)d_in[36];
  const float* aW2 = (const float*)d_in[37]; const float* ab2 = (const float*)d_in[38];
  const float* ag0b = (const float*)d_in[39]; const float* ag0w = (const float*)d_in[40];
  const float* ag1b = (const float*)d_in[41]; const float* ag1w = (const float*)d_in[42];

  const float* gW0 = (const float*)d_in[43]; const float* gW1 = (const float*)d_in[44];
  const float* gW2 = (const float*)d_in[45]; const float* gb2 = (const float*)d_in[46];
  const float* gg0b = (const float*)d_in[47]; const float* gg0w = (const float*)d_in[48];
  const float* gg1b = (const float*)d_in[49]; const float* gg1w = (const float*)d_in[50];

  const float* lW0 = (const float*)d_in[51]; const float* lW1 = (const float*)d_in[52];
  const float* lW2 = (const float*)d_in[53]; const float* lb2 = (const float*)d_in[54];
  const float* lg0b = (const float*)d_in[55]; const float* lg0w = (const float*)d_in[56];
  const float* lg1b = (const float*)d_in[57]; const float* lg1w = (const float*)d_in[58];

  // ---- workspace layout (bytes, 256B-aligned) ----
  char* ws = (char*)d_ws;
  size_t off = 0;
  auto alloc = [&](size_t bytes) { char* p = ws + off; off += (bytes + 255) & ~(size_t)255; return p; };
  u16* wsLW0 = (u16*)alloc((size_t)512 * 128 * 2);
  u16* wsLW1 = (u16*)alloc((size_t)512 * 512 * 2);
  u16* wsLW2 = (u16*)alloc((size_t)512 * 512 * 2);
  u16* wsGW0 = (u16*)alloc((size_t)256 * 512 * 2);
  u16* wsGW1 = (u16*)alloc((size_t)256 * 256 * 2);
  u16* wsGW2 = (u16*)alloc((size_t)256 * 256 * 2);
  u16* wsAW0 = (u16*)alloc((size_t)256 * 256 * 2);
  u16* wsAW1 = (u16*)alloc((size_t)256 * 256 * 2);
  u16* featbf = (u16*)alloc((size_t)N * 96 * 2);
  float* agg  = (float*)alloc((size_t)N * 512 * 4);
  float* xg   = (float*)alloc((size_t)N * 256 * 4);
  float* gate = (float*)alloc((size_t)N * 4);
  float* ebuf = (float*)alloc((size_t)N * 4);
  float* gmax = (float*)alloc((size_t)B * 4);
  float* denom = (float*)alloc((size_t)B * 4);
  float* pooled = (float*)alloc((size_t)B * 256 * 4);
  (void)ws_size;

  // ---- K0: weight conversion to bf16 ----
  auto cvt = [&](const float* s, u16* d, int rows, int cs, int cd) {
    long tot = (long)rows * cd;
    convert_w_kernel<<<(int)((tot + 255) / 256), 256, 0, stream>>>(s, d, rows, cs, cd);
  };
  cvt(lW0, wsLW0, 512, 100, 128);
  cvt(lW1, wsLW1, 512, 512, 512);
  cvt(lW2, wsLW2, 512, 512, 512);
  cvt(gW0, wsGW0, 256, 512, 512);
  cvt(gW1, wsGW1, 256, 256, 256);
  cvt(gW2, wsGW2, 256, 256, 256);
  cvt(aW0, wsAW0, 256, 256, 256);
  cvt(aW1, wsAW1, 256, 256, 256);

  // ---- K1: encoders ----
  EncParams EP{ x, pos,
                eW0, eW1, eW2, eb0, eb1, eb2, eg0w, eg0b, eg1w, eg1b,
                pW0, pW1, pW2, pb0, pb1, pb2, pg0w, pg0b, pg1w, pg1b,
                featbf, N };
  encode_kernel<<<(N + 255) / 256, 256, 0, stream>>>(EP);

  // ---- zero accumulation buffers ----
  hipMemsetAsync(agg, 0, (size_t)N * 512 * 4, stream);
  hipMemsetAsync(denom, 0, (size_t)B * 4, stream);
  hipMemsetAsync(pooled, 0, (size_t)B * 256 * 4, stream);
  init_gmax_kernel<<<1, 32, 0, stream>>>(gmax);

  // ---- K2: local_nn + scatter-add (32 rows/block) ----
  {
    LocalParams LP{ pos, ei, featbf, wsLW0, wsLW1, wsLW2,
                    lb2, lg0w, lg0b, lg1w, lg1b, agg, N, E };
    long rows = (long)E + N;
    local_nn_kernel<<<(int)((rows + 31) / 32), 256, 0, stream>>>(LP);
  }

  // ---- K3: global_nn (32 rows/block) ----
  {
    GlobParams GP{ agg, wsGW0, wsGW1, wsGW2, gb2, gg0w, gg0b, gg1w, gg1b, xg, N };
    global_nn_kernel<<<(N + 31) / 32, 256, 0, stream>>>(GP);
  }

  // ---- K4: gate_nn (32 rows/block) ----
  {
    GateParams AP{ xg, wsAW0, wsAW1, aW2, ab2, ag0w, ag0b, ag1w, ag1b, gate, N };
    gate_nn_kernel<<<(N + 31) / 32, 256, 0, stream>>>(AP);
  }

  // ---- K5: attention pooling ----
  gate_max_kernel<<<(N + 255) / 256, 256, 0, stream>>>(gate, batch, gmax, N);
  gate_exp_kernel<<<(N + 255) / 256, 256, 0, stream>>>(gate, batch, gmax, ebuf, denom, N);
  pool_kernel<<<(int)(((long)N * 256 + 255) / 256), 256, 0, stream>>>(
      xg, batch, ebuf, denom, pooled, N);

  // ---- K6: decoder -> d_out [16,6] ----
  DecParams DP{ pooled, dW0, dW1, dW2, db0, db1, db2, dg0w, dg0b, dg1w, dg1b,
                (float*)d_out };
  dec_kernel<<<1, 256, 0, stream>>>(DP);
}
